// GeneratedTreeClassifier_1425929142327
// MI455X (gfx1250) — compile-verified, matrix operand order checked by reading, not run on hardware
//
#include <hip/hip_runtime.h>
#include <hip/hip_bf16.h>
#include <math.h>

// ---------------- problem constants (from reference) ----------------
constexpr int T_ = 64;    // trees
constexpr int N_ = 63;    // internal nodes per tree (2^6 - 1)
constexpr int D_ = 512;   // feature dim
constexpr int L_ = 64;    // leaves per tree
constexpr int C_ = 100;   // classes
constexpr int CP = 128;   // classes padded to 8 n-tiles of 16

// ---------------- tiling ----------------
#define BM 32            // rows of x per workgroup
#define BN 64            // padded node dim (one tree)
#define BK 64            // K chunk for GEMM1
#define XS 72            // bf16 LDS row stride (pad -> conflict-free b128 reads)
#define DS 68            // f32 LDS row stride for dec

typedef __attribute__((ext_vector_type(16))) __bf16 bf16x16;
typedef __attribute__((ext_vector_type(8)))  float  v8f;

// ---------------- bf16 split helpers (round-to-nearest-even) ----------------
__device__ __forceinline__ unsigned short f2bf(float f) {
    unsigned int u = __float_as_uint(f);
    unsigned int r = u + 0x7FFFu + ((u >> 16) & 1u);
    return (unsigned short)(r >> 16);
}
__device__ __forceinline__ float bf2f(unsigned short h) {
    return __uint_as_float(((unsigned int)h) << 16);
}
__device__ __forceinline__ void split2(float v, unsigned short& h, unsigned short& l) {
    h = f2bf(v);
    l = f2bf(v - bf2f(h));
}
// split a float4 and store 4 hi-bf16 + 4 lo-bf16 (packed dword stores)
__device__ __forceinline__ void split_store4(unsigned short* hb, unsigned short* lb,
                                             int off, float4 v) {
    unsigned short h0,h1,h2,h3,l0,l1,l2,l3;
    split2(v.x,h0,l0); split2(v.y,h1,l1); split2(v.z,h2,l2); split2(v.w,h3,l3);
    *(unsigned int*)(hb + off    ) = (unsigned)h0 | ((unsigned)h1 << 16);
    *(unsigned int*)(hb + off + 2) = (unsigned)h2 | ((unsigned)h3 << 16);
    *(unsigned int*)(lb + off    ) = (unsigned)l0 | ((unsigned)l1 << 16);
    *(unsigned int*)(lb + off + 2) = (unsigned)l2 | ((unsigned)l3 << 16);
}

// 16-bit A/B fragment of a 16x32 (MxK / NxK) tile, K-contiguous row-major.
// ISA layout: lanes 0-15 & 16-31 both hold rows 0-15; VGPR0-3 = K[g*8..g*8+7],
// VGPR4-7 = K[16+g*8..16+g*8+7]  -> two 16B reads per lane.
__device__ __forceinline__ bf16x16 load_frag(const unsigned short* buf, int row,
                                             int stride, int ks, int g) {
    union { uint4 q[2]; bf16x16 v; } u;
    const unsigned short* p = buf + row * stride + ks + g * 8;
    u.q[0] = *(const uint4*)(p);
    u.q[1] = *(const uint4*)(p + 16);
    return u.v;
}

// ---- kernel 1: PT_hi/PT_lo[t][c][l] = split_bf16( tw[t]*softmax(ll[t,l])[c] ) ----
__global__ __launch_bounds__(128) void prep_P(const float* __restrict__ ll,
                                              const float* __restrict__ tw,
                                              unsigned short* __restrict__ PTh,
                                              unsigned short* __restrict__ PTl) {
    __shared__ float red[128];
    const int row = blockIdx.x;           // t*64 + l
    const int t   = row >> 6;
    const int lf  = row & 63;
    const int c   = threadIdx.x;
    float v = (c < C_) ? ll[(size_t)row * C_ + c] : -INFINITY;
    red[c] = v; __syncthreads();
    for (int s = 64; s > 0; s >>= 1) {
        if (c < s) red[c] = fmaxf(red[c], red[c + s]);
        __syncthreads();
    }
    float mx = red[0]; __syncthreads();
    float e = (c < C_) ? __expf(v - mx) : 0.f;
    red[c] = e; __syncthreads();
    for (int s = 64; s > 0; s >>= 1) {
        if (c < s) red[c] += red[c + s];
        __syncthreads();
    }
    float sum = red[0];
    float p = (c < C_) ? (tw[t] * e / sum) : 0.f;
    unsigned short h, lo;
    split2(p, h, lo);
    // transposed, K(leaf)-contiguous layout for direct B-fragment loads
    size_t idx = ((size_t)t * CP + c) * L_ + lf;
    PTh[idx] = h;
    PTl[idx] = lo;
}

// ---------------- kernel 2: fused trees ----------------
__global__ __launch_bounds__(256) void tree_fused(const float* __restrict__ x,
                                                  const float* __restrict__ sw,
                                                  const float* __restrict__ sb,
                                                  const unsigned short* __restrict__ PTh,
                                                  const unsigned short* __restrict__ PTl,
                                                  float* __restrict__ out) {
    __shared__ unsigned short xh[BM * XS], xl[BM * XS];
    __shared__ unsigned short wh[BN * XS], wl[BN * XS];
    __shared__ float dec[BM * DS];
    __shared__ unsigned short lph[BM * XS], lpl[BM * XS];

    const int tid  = threadIdx.x;
    const int lane = tid & 31;
    const int wid  = tid >> 5;          // 8 waves
    const int g    = lane >> 4;         // half-wave
    const int ln   = lane & 15;
    const int blockRow = blockIdx.x * BM;

    const int m1 = wid & 1;             // M tile (2)
    const int n1 = wid >> 1;            // N tile for GEMM1 (4)
    const int ntA = (wid >> 1) * 2;     // N tiles for GEMM2 output (8 over CP=128)
    const int ntB = ntA + 1;

    v8f out0 = {}; v8f out1 = {};       // persistent per-wave output accumulators

    for (int t = 0; t < T_; ++t) {
        // ---------- GEMM1: dec_logits = x_tile (32x512) . W_t^T (512x64) ----------
        v8f acc = {};
        for (int kc = 0; kc < D_ / BK; ++kc) {
            // stage x chunk (32x64 fp32 -> bf16 hi/lo)
            #pragma unroll
            for (int i = 0; i < (BM * BK / 4) / 256; ++i) {   // 2
                int f = tid + i * 256;
                int row = f >> 4, kv = f & 15;
                float4 v = *(const float4*)(x + (size_t)(blockRow + row) * D_
                                              + kc * BK + kv * 4);
                split_store4(xh, xl, row * XS + kv * 4, v);
            }
            // stage W chunk (64x64 fp32 -> bf16 hi/lo; row 63 zero-padded)
            #pragma unroll
            for (int i = 0; i < (BN * BK / 4) / 256; ++i) {   // 4
                int f = tid + i * 256;
                int n = f >> 4, kv = f & 15;
                float4 v = make_float4(0.f, 0.f, 0.f, 0.f);
                if (n < N_)
                    v = *(const float4*)(sw + ((size_t)t * N_ + n) * D_
                                            + kc * BK + kv * 4);
                split_store4(wh, wl, n * XS + kv * 4, v);
            }
            __syncthreads();
            #pragma unroll
            for (int ks = 0; ks < BK; ks += 32) {
                bf16x16 Ah = load_frag(xh, m1 * 16 + ln, XS, ks, g);
                bf16x16 Al = load_frag(xl, m1 * 16 + ln, XS, ks, g);
                bf16x16 Bh = load_frag(wh, n1 * 16 + ln, XS, ks, g);
                bf16x16 Bl = load_frag(wl, n1 * 16 + ln, XS, ks, g);
                // split-bf16: hi*hi + hi*lo + lo*hi  ~= fp32 product
                acc = __builtin_amdgcn_wmma_f32_16x16x32_bf16(false, Ah, false, Bh,
                        (short)0, acc, false, false);
                acc = __builtin_amdgcn_wmma_f32_16x16x32_bf16(false, Ah, false, Bl,
                        (short)0, acc, false, false);
                acc = __builtin_amdgcn_wmma_f32_16x16x32_bf16(false, Al, false, Bh,
                        (short)0, acc, false, false);
            }
            __syncthreads();
        }
        // ---------- bias + sigmoid -> dec in LDS ----------
        {
            int node = n1 * 16 + ln;
            float bias = (node < N_) ? sb[t * N_ + node] : 0.f;
            #pragma unroll
            for (int r = 0; r < 8; ++r) {
                int m = r + g * 8;                       // C/D layout: lanes16-31 -> M+8
                float z = acc[r] + bias;                 // TEMPERATURE == 1.0
                dec[(m1 * 16 + m) * DS + node] = 1.f / (1.f + __expf(-z));
            }
        }
        __syncthreads();
        // ---------- leaf path probabilities -> split bf16 hi/lo in LDS ----------
        #pragma unroll
        for (int i = 0; i < (BM * L_) / 256; ++i) {      // 8
            int e = tid + i * 256;
            int b = e >> 6, lf = e & 63;
            float p = 1.f;
            #pragma unroll
            for (int d = 0; d < 6; ++d) {
                int node = (1 << d) - 1 + (lf >> (6 - d));
                int bit  = (lf >> (5 - d)) & 1;
                float dv = dec[b * DS + node];
                p *= bit ? dv : (1.f - dv);
            }
            unsigned short h, lo;
            split2(p, h, lo);
            lph[b * XS + lf] = h;
            lpl[b * XS + lf] = lo;
        }
        __syncthreads();
        // ---------- GEMM2: out += lp (32x64) . P_t^T-frags (64x128), bf16 WMMA ----
        {
            const unsigned short* Bh0 = PTh + ((size_t)t * CP + ntA * 16 + ln) * L_;
            const unsigned short* Bl0 = PTl + ((size_t)t * CP + ntA * 16 + ln) * L_;
            const unsigned short* Bh1 = PTh + ((size_t)t * CP + ntB * 16 + ln) * L_;
            const unsigned short* Bl1 = PTl + ((size_t)t * CP + ntB * 16 + ln) * L_;
            #pragma unroll
            for (int ks = 0; ks < L_; ks += 32) {
                bf16x16 Ah = load_frag(lph, m1 * 16 + ln, XS, ks, g);
                bf16x16 Al = load_frag(lpl, m1 * 16 + ln, XS, ks, g);
                bf16x16 BAh = load_frag(Bh0, 0, 0, ks, g);
                bf16x16 BAl = load_frag(Bl0, 0, 0, ks, g);
                bf16x16 BBh = load_frag(Bh1, 0, 0, ks, g);
                bf16x16 BBl = load_frag(Bl1, 0, 0, ks, g);
                out0 = __builtin_amdgcn_wmma_f32_16x16x32_bf16(false, Ah, false, BAh,
                        (short)0, out0, false, false);
                out0 = __builtin_amdgcn_wmma_f32_16x16x32_bf16(false, Ah, false, BAl,
                        (short)0, out0, false, false);
                out0 = __builtin_amdgcn_wmma_f32_16x16x32_bf16(false, Al, false, BAh,
                        (short)0, out0, false, false);
                out1 = __builtin_amdgcn_wmma_f32_16x16x32_bf16(false, Ah, false, BBh,
                        (short)0, out1, false, false);
                out1 = __builtin_amdgcn_wmma_f32_16x16x32_bf16(false, Ah, false, BBl,
                        (short)0, out1, false, false);
                out1 = __builtin_amdgcn_wmma_f32_16x16x32_bf16(false, Al, false, BBh,
                        (short)0, out1, false, false);
            }
        }
        __syncthreads();
    }
    // ---------- store output tile ----------
    #pragma unroll
    for (int r = 0; r < 8; ++r) {
        int m   = r + g * 8;
        int row = blockRow + m1 * 16 + m;
        int cA  = ntA * 16 + ln;
        int cB  = ntB * 16 + ln;
        if (cA < C_) out[(size_t)row * C_ + cA] = out0[r];
        if (cB < C_) out[(size_t)row * C_ + cB] = out1[r];
    }
}

// ---------------- launcher ----------------
extern "C" void kernel_launch(void* const* d_in, const int* in_sizes, int n_in,
                              void* d_out, int out_size, void* d_ws, size_t ws_size,
                              hipStream_t stream) {
    const float* x  = (const float*)d_in[0];   // (B, 512)
    const float* sw = (const float*)d_in[1];   // (64, 63, 512)
    const float* sb = (const float*)d_in[2];   // (64, 63)
    const float* ll = (const float*)d_in[3];   // (64, 64, 100)
    const float* tw = (const float*)d_in[4];   // (64,)
    float* out = (float*)d_out;                // (B, 100)

    // workspace: PT_hi (1 MB) then PT_lo (1 MB), bf16, [t][c(128)][l(64)]
    unsigned short* PTh = (unsigned short*)d_ws;
    unsigned short* PTl = PTh + (size_t)T_ * CP * L_;

    const int B = in_sizes[0] / D_;            // 8192

    prep_P<<<T_ * L_, 128, 0, stream>>>(ll, tw, PTh, PTl);
    tree_fused<<<B / BM, 256, 0, stream>>>(x, sw, sb, PTh, PTl, out);
    (void)n_in; (void)out_size; (void)ws_size;
}
